// Block_71554155151924
// MI455X (gfx1250) — matrix operand order repaired and checked by reading, not compiled, not run on hardware
//
#include <hip/hip_runtime.h>
#include <hip/hip_bf16.h>

typedef _Float16 h16;
typedef __attribute__((ext_vector_type(16))) _Float16 v16h;
typedef __attribute__((ext_vector_type(2)))  _Float16 h2;
typedef __attribute__((ext_vector_type(8)))  float    v8f;

union vfrag {
  v16h v;
  h2   p[8];
};

#define B_   2
#define T_   2048
#define C_   1024
#define H_   16
#define DK_  64
#define FF_  4096
#define NROW (B_ * T_)          // 4096
#define QKVW (H_ * 3 * DK_)     // 3072

// Pair K-map: fragment halves (2p,2p+1) sit at k = ((p&4)<<2) + 8*g + 2*(p&3).
// For p=0..3 / p=4..7 each half-fragment is 8 contiguous halves -> ds_load_b128.
__device__ __forceinline__ int kpair(int p, int g) {
  return ((p & 4) << 2) + g * 8 + ((p & 3) << 1);
}

// ---------------------------------------------------------------------------
// Elementwise converts
// ---------------------------------------------------------------------------
__global__ void k_f32_to_f16(const float* __restrict__ src, h16* __restrict__ dst, int n) {
  int i = (blockIdx.x * blockDim.x + threadIdx.x) * 4;
  if (i < n) {
    float4 f = *(const float4*)(src + i);
    dst[i + 0] = (h16)f.x;
    dst[i + 1] = (h16)f.y;
    dst[i + 2] = (h16)f.z;
    dst[i + 3] = (h16)f.w;
  }
}

// f32 [K,N] row-major -> f16 transposed [N,K]: B-fragment loads become
// contiguous in K (b128 LDS reads, no repacking).
__global__ void k_f32_to_f16_T(const float* __restrict__ src, h16* __restrict__ dst,
                               int K, int N) {
  int idx = blockIdx.x * 256 + threadIdx.x;
  if (idx >= K * N) return;
  int k = idx / N, n = idx % N;           // coalesced read, strided write
  dst[(size_t)n * K + k] = (h16)src[idx];
}

// Wqkv [H, C, 3*DK] f32 -> transposed f16 [H*3*DK, C] (row = h*192+f, col = c)
__global__ void k_pack_wqkv_T(const float* __restrict__ src, h16* __restrict__ dst) {
  int idx = blockIdx.x * 256 + threadIdx.x;
  if (idx >= H_ * C_ * 3 * DK_) return;
  int h = idx / (C_ * 3 * DK_);
  int rem = idx % (C_ * 3 * DK_);
  int c = rem / (3 * DK_);
  int f = rem % (3 * DK_);
  dst[(size_t)(h * (3 * DK_) + f) * C_ + c] = (h16)src[idx];
}

// ---------------------------------------------------------------------------
// Tiled WMMA GEMM: C[M,N] = A[M,K] * Bt[N,K]^T, f16 in, f32 accumulate.
// Block tile 128x128, K-step 32, 256 threads = 8 waves (4x2), wave tile 32x64.
// Double-buffered LDS with register staging; both tiles identical 128x32 shape.
// EPI 0: QKV scatter (+bias, Q scaled by 1/sqrt(dk), V written transposed)
// EPI 1: bias + ReLU -> f16
// EPI 2: bias + residual -> f32
// ---------------------------------------------------------------------------
#define BM 128
#define BN 128
#define BK 32
#define LDS_S 40   // padded LDS stride (halves): 20 dwords -> conflict-free rows

template <int EPI>
__global__ __launch_bounds__(256) void k_gemm(
    const h16* __restrict__ A, const h16* __restrict__ Bt, int M, int N, int K,
    const float* __restrict__ bias, const float* __restrict__ res,
    float* __restrict__ outf, h16* __restrict__ outh,
    h16* __restrict__ Qo, h16* __restrict__ Ko, h16* __restrict__ Vo) {
  __shared__ __align__(16) h16 sA[2][BM * LDS_S];
  __shared__ __align__(16) h16 sB[2][BN * LDS_S];

  const int tid  = threadIdx.x;
  const int bm   = blockIdx.y * BM;
  const int bn   = blockIdx.x * BN;
  const int wave = tid >> 5;
  const int lane = tid & 31;
  const int wm   = (wave & 3) * 32;
  const int wn   = (wave >> 2) * 64;
  const int ln   = lane & 15;
  const int g    = lane >> 4;

  // Cooperative tile chunk: 512 uint4 per tile / 256 threads = 2 each.
  const int row0 = tid >> 2,         c80 = (tid & 3) * 8;
  const int row1 = (256 + tid) >> 2, c81 = c80;

  const v8f vzero = {0.f, 0.f, 0.f, 0.f, 0.f, 0.f, 0.f, 0.f};
  v8f acc[2][4];
#pragma unroll
  for (int i = 0; i < 2; ++i)
#pragma unroll
    for (int j = 0; j < 4; ++j) acc[i][j] = vzero;

  uint4 ra0, ra1, rb0, rb1;
  auto gload = [&](int k0) {
    ra0 = *(const uint4*)(A  + (size_t)(bm + row0) * K + k0 + c80);
    ra1 = *(const uint4*)(A  + (size_t)(bm + row1) * K + k0 + c81);
    rb0 = *(const uint4*)(Bt + (size_t)(bn + row0) * K + k0 + c80);
    rb1 = *(const uint4*)(Bt + (size_t)(bn + row1) * K + k0 + c81);
  };
  auto lstore = [&](int buf) {
    *(uint4*)(&sA[buf][row0 * LDS_S + c80]) = ra0;
    *(uint4*)(&sA[buf][row1 * LDS_S + c81]) = ra1;
    *(uint4*)(&sB[buf][row0 * LDS_S + c80]) = rb0;
    *(uint4*)(&sB[buf][row1 * LDS_S + c81]) = rb1;
  };

  gload(0);
  lstore(0);
  __syncthreads();

  int buf = 0;
  for (int k0 = 0; k0 < K; k0 += BK) {
    const int kn = k0 + BK;
    if (kn < K) gload(kn);                           // overlap with WMMA below
    if (kn + BK < K) {                               // L2 warmup, distance 2
      __builtin_prefetch(A  + (size_t)(bm + row0) * K + kn + BK + c80, 0, 1);
      __builtin_prefetch(Bt + (size_t)(bn + row0) * K + kn + BK + c80, 0, 1);
    }

    vfrag af[2];
#pragma unroll
    for (int i = 0; i < 2; ++i) {
      int r = wm + i * 16 + ln;
#pragma unroll
      for (int p = 0; p < 8; ++p)
        af[i].p[p] = *(const h2*)&sA[buf][r * LDS_S + kpair(p, g)];
    }
#pragma unroll
    for (int jn = 0; jn < 4; ++jn) {
      vfrag bf;
      int cc = wn + jn * 16 + ln;
#pragma unroll
      for (int p = 0; p < 8; ++p)
        bf.p[p] = *(const h2*)&sB[buf][cc * LDS_S + kpair(p, g)];
#pragma unroll
      for (int i = 0; i < 2; ++i)
        acc[i][jn] = __builtin_amdgcn_wmma_f32_16x16x32_f16(
            false, af[i].v, false, bf.v, (short)0, acc[i][jn], false, false);
    }

    if (kn < K) lstore(buf ^ 1);
    __syncthreads();
    buf ^= 1;
  }

  // Epilogue. C/D layout: row M = v + 8*g, col = lane&15.
#pragma unroll
  for (int i = 0; i < 2; ++i)
#pragma unroll
    for (int jn = 0; jn < 4; ++jn)
#pragma unroll
      for (int v = 0; v < 8; ++v) {
        int r = bm + wm + i * 16 + v + 8 * g;
        int c = bn + wn + jn * 16 + ln;
        float val = acc[i][jn][v];
        if (EPI == 0) {
          val += bias[c];  // bqkv flat [H*192] == column index
          int h = c / (3 * DK_), f = c % (3 * DK_);
          int b = r >> 11, t = r & (T_ - 1);
          if (f < DK_) {
            Qo[(((size_t)(b * H_ + h)) * T_ + t) * DK_ + f] = (h16)(val * 0.125f);
          } else if (f < 2 * DK_) {
            Ko[(((size_t)(b * H_ + h)) * T_ + t) * DK_ + (f - DK_)] = (h16)val;
          } else {  // V transposed: [B*H, DK, T]
            Vo[(((size_t)(b * H_ + h)) * DK_ + (f - 2 * DK_)) * T_ + t] = (h16)val;
          }
        } else if (EPI == 1) {
          val += bias[c];
          outh[(size_t)r * N + c] = (h16)(val > 0.f ? val : 0.f);
        } else {
          outf[(size_t)r * N + c] = val + bias[c] + res[(size_t)r * N + c];
        }
      }
}

// ---------------------------------------------------------------------------
// Causal flash attention.
// Q,K f16 [B*H, T, 64] (Q pre-scaled by 1/sqrt(dk)); V f16 transposed [B*H, 64, T].
// 128 threads = 4 waves; block owns 64 query rows, wave owns 16.
// Key tiles of 32 staged cooperatively in LDS, double buffered.
// Causal mask evaluated only for the <=2 diagonal blocks per wave (wave-uniform
// branch); softmax row-sum accumulated by a 5th WMMA against an all-ones B
// fragment (same rescale recurrence as O), so only the row-max needs shuffles.
// ---------------------------------------------------------------------------
#define KS 72   // sK row stride (halves)
#define VS 40   // sV row stride (halves)

__global__ __launch_bounds__(128) void k_attn(
    const h16* __restrict__ Q, const h16* __restrict__ Kt,
    const h16* __restrict__ Vt, h16* __restrict__ Oh) {
  __shared__ __align__(16) h16 sK[2][32 * KS];   // [key][dk]
  __shared__ __align__(16) h16 sV[2][64 * VS];   // [dk][key]
  __shared__ __align__(16) h16 sP[4][16 * VS];   // per-wave P tile

  const int bh    = blockIdx.y;
  const int tid   = threadIdx.x;
  const int wave  = tid >> 5;
  const int lane  = tid & 31;
  const int ln    = lane & 15;
  const int g     = lane >> 4;
  const int b     = bh >> 4;
  const int h     = bh & (H_ - 1);
  const size_t hb = (size_t)bh * T_ * DK_;
  const int qbase = blockIdx.x * 64 + wave * 16;
  const int kend  = blockIdx.x * 64 + 64;        // uniform across waves

  // K tile: 32x64 halves = 256 uint4; V tile: 64x32 halves = 256 uint4.
  const int krow0 = tid >> 2,         kc0 = (tid & 3) * 8;
  const int krow1 = (128 + tid) >> 2, kc1 = kc0;

  uint4 rk0, rk1, rv0, rv1;
  auto gloadKV = [&](int kb) {
    rk0 = *(const uint4*)(Kt + hb + (size_t)(kb + (tid >> 3)) * DK_ + (tid & 7) * 8);
    rk1 = *(const uint4*)(Kt + hb + (size_t)(kb + 16 + (tid >> 3)) * DK_ + (tid & 7) * 8);
    rv0 = *(const uint4*)(Vt + hb + (size_t)krow0 * T_ + kb + kc0);
    rv1 = *(const uint4*)(Vt + hb + (size_t)krow1 * T_ + kb + kc1);
  };
  auto lstoreKV = [&](int buf) {
    *(uint4*)(&sK[buf][(tid >> 3) * KS + (tid & 7) * 8]) = rk0;
    *(uint4*)(&sK[buf][(16 + (tid >> 3)) * KS + (tid & 7) * 8]) = rk1;
    *(uint4*)(&sV[buf][krow0 * VS + kc0]) = rv0;
    *(uint4*)(&sV[buf][krow1 * VS + kc1]) = rv1;
  };

  // Q fragments over dk=64 -> two K-steps of 32, paired global loads.
  vfrag aq[2];
#pragma unroll
  for (int kk = 0; kk < 2; ++kk)
#pragma unroll
    for (int p = 0; p < 8; ++p)
      aq[kk].p[p] = *(const h2*)&Q[hb + (size_t)(qbase + ln) * DK_ + kk * 32 + kpair(p, g)];

  // All-ones B fragment for row-sum accumulation.
  vfrag ones;
#pragma unroll
  for (int j = 0; j < 16; ++j) ones.v[j] = (h16)1.f;

  const v8f vzero = {0.f, 0.f, 0.f, 0.f, 0.f, 0.f, 0.f, 0.f};
  v8f o[4], osum = vzero;
#pragma unroll
  for (int nt = 0; nt < 4; ++nt) o[nt] = vzero;
  float rmax[8];
#pragma unroll
  for (int v = 0; v < 8; ++v) rmax[v] = -1e30f;

  h16* P = &sP[wave][0];

  gloadKV(0);
  lstoreKV(0);
  __syncthreads();

  int buf = 0;
  for (int kb = 0; kb < kend; kb += 32) {
    const int knext = kb + 32;
    if (knext < kend) gloadKV(knext);

    // logits: 16 rows x 32 keys = two 16x16 C fragments
    v8f s[2];
    s[0] = vzero;
    s[1] = vzero;
#pragma unroll
    for (int jn = 0; jn < 2; ++jn)
#pragma unroll
      for (int kk = 0; kk < 2; ++kk) {
        vfrag bk;
#pragma unroll
        for (int p = 0; p < 8; ++p)
          bk.p[p] = *(const h2*)&sK[buf][(jn * 16 + ln) * KS + kk * 32 + kpair(p, g)];
        s[jn] = __builtin_amdgcn_wmma_f32_16x16x32_f16(
            false, aq[kk].v, false, bk.v, (short)0, s[jn], false, false);
      }

    // Causal mask only on the (<=2) diagonal blocks of this wave.
    const bool masked = (kb + 31 > qbase);   // wave-uniform scalar branch
    if (masked) {
#pragma unroll
      for (int v = 0; v < 8; ++v) {
        int r = qbase + v + 8 * g;
        if (kb + ln > r)      s[0][v] = -1e30f;
        if (kb + 16 + ln > r) s[1][v] = -1e30f;
      }
    }

    // Online softmax: row max via shfl ladder; sums handled by WMMA below.
#pragma unroll
    for (int v = 0; v < 8; ++v) {
      float t = fmaxf(s[0][v], s[1][v]);
      for (int off = 1; off < 16; off <<= 1) t = fmaxf(t, __shfl_xor(t, off, 32));
      float mnew  = fmaxf(rmax[v], t);
      float scale = __expf(rmax[v] - mnew);
      s[0][v] = __expf(s[0][v] - mnew);
      s[1][v] = __expf(s[1][v] - mnew);
      rmax[v] = mnew;
      osum[v] *= scale;
#pragma unroll
      for (int nt = 0; nt < 4; ++nt) o[nt][v] *= scale;
    }

    // Stage P (C-layout) through LDS into A-fragment layout.
#pragma unroll
    for (int v = 0; v < 8; ++v) {
      P[(v + 8 * g) * VS + ln]      = (h16)s[0][v];
      P[(v + 8 * g) * VS + 16 + ln] = (h16)s[1][v];
    }
    vfrag ap;
#pragma unroll
    for (int p = 0; p < 8; ++p)
      ap.p[p] = *(const h2*)&P[ln * VS + kpair(p, g)];

    // Row sums: osum += P @ ones (every column holds the row sum).
    osum = __builtin_amdgcn_wmma_f32_16x16x32_f16(
        false, ap.v, false, ones.v, (short)0, osum, false, false);

    // O += P @ V  (sV is [dk][key]: columns of V are K-contiguous rows)
#pragma unroll
    for (int nt = 0; nt < 4; ++nt) {
      vfrag bv;
#pragma unroll
      for (int p = 0; p < 8; ++p)
        bv.p[p] = *(const h2*)&sV[buf][(nt * 16 + ln) * VS + kpair(p, g)];
      o[nt] = __builtin_amdgcn_wmma_f32_16x16x32_f16(
          false, ap.v, false, bv.v, (short)0, o[nt], false, false);
    }

    if (knext < kend) lstoreKV(buf ^ 1);
    __syncthreads();
    buf ^= 1;
  }

#pragma unroll
  for (int v = 0; v < 8; ++v) {
    int r = qbase + v + 8 * g;
    float inv = 1.f / osum[v];   // lane-local: all columns carry the row sum
#pragma unroll
    for (int nt = 0; nt < 4; ++nt)
      Oh[((size_t)(b * T_ + r)) * C_ + h * DK_ + nt * 16 + ln] = (h16)(o[nt][v] * inv);
  }
}

// ---------------------------------------------------------------------------
// LayerNorm: one block per row of 1024, writes f32 (+ optional f16 copy).
// ---------------------------------------------------------------------------
__global__ __launch_bounds__(256) void k_layernorm(
    const float* __restrict__ in, const float* __restrict__ gamma,
    const float* __restrict__ beta, float* __restrict__ outf,
    h16* __restrict__ outh) {
  __shared__ float s1[256], s2[256];
  const int row = blockIdx.x;
  const float* x = in + (size_t)row * C_;
  float4 f = *(const float4*)(x + threadIdx.x * 4);
  s1[threadIdx.x] = f.x + f.y + f.z + f.w;
  s2[threadIdx.x] = f.x * f.x + f.y * f.y + f.z * f.z + f.w * f.w;
  __syncthreads();
  for (int sft = 128; sft > 0; sft >>= 1) {
    if (threadIdx.x < sft) {
      s1[threadIdx.x] += s1[threadIdx.x + sft];
      s2[threadIdx.x] += s2[threadIdx.x + sft];
    }
    __syncthreads();
  }
  float mean = s1[0] * (1.f / C_);
  float var  = s2[0] * (1.f / C_) - mean * mean;
  float rstd = rsqrtf(var + 1e-5f);
  float vals[4] = {f.x, f.y, f.z, f.w};
#pragma unroll
  for (int i = 0; i < 4; ++i) {
    int c = threadIdx.x * 4 + i;
    float nv = (vals[i] - mean) * rstd * gamma[c] + beta[c];
    outf[(size_t)row * C_ + c] = nv;
    if (outh) outh[(size_t)row * C_ + c] = (h16)nv;
  }
}

// ---------------------------------------------------------------------------
extern "C" void kernel_launch(void* const* d_in, const int* in_sizes, int n_in,
                              void* d_out, int out_size, void* d_ws, size_t ws_size,
                              hipStream_t stream) {
  (void)in_sizes; (void)n_in; (void)out_size; (void)ws_size;
  const float* x    = (const float*)d_in[0];
  const float* Wqkv = (const float*)d_in[1];
  const float* bqkv = (const float*)d_in[2];
  const float* WO   = (const float*)d_in[3];
  const float* bO   = (const float*)d_in[4];
  const float* g1   = (const float*)d_in[5];
  const float* be1  = (const float*)d_in[6];
  const float* g2   = (const float*)d_in[7];
  const float* be2  = (const float*)d_in[8];
  const float* W1   = (const float*)d_in[9];
  const float* b1   = (const float*)d_in[10];
  const float* W2   = (const float*)d_in[11];
  const float* b2   = (const float*)d_in[12];
  float* out = (float*)d_out;

  char* p = (char*)d_ws;
  auto alloc = [&](size_t bytes) {
    char* r = p;
    p += (bytes + 255) & ~(size_t)255;
    return r;
  };
  h16*   xh    = (h16*)alloc((size_t)NROW * C_ * 2);     // reused as attn output
  h16*   WqkvT = (h16*)alloc((size_t)C_ * QKVW * 2);     // [3072, C]
  h16*   WOT   = (h16*)alloc((size_t)C_ * C_ * 2);       // [C, C]
  h16*   W1T   = (h16*)alloc((size_t)C_ * FF_ * 2);      // [FF, C]
  h16*   W2T   = (h16*)alloc((size_t)FF_ * C_ * 2);      // [C, FF]
  h16*   Qh    = (h16*)alloc((size_t)B_ * H_ * T_ * DK_ * 2);
  h16*   Kh    = (h16*)alloc((size_t)B_ * H_ * T_ * DK_ * 2);
  h16*   Vth   = (h16*)alloc((size_t)B_ * H_ * T_ * DK_ * 2);  // transposed [BH,DK,T]
  float* y1    = (float*)alloc((size_t)NROW * C_ * 4);   // reused as y2
  float* x1    = (float*)alloc((size_t)NROW * C_ * 4);
  h16*   x1h   = (h16*)alloc((size_t)NROW * C_ * 2);
  h16*   h1h   = (h16*)alloc((size_t)NROW * FF_ * 2);
  h16*   attnout = xh;
  float* y2      = y1;

  int n;
  n = NROW * C_;
  k_f32_to_f16<<<(n / 4 + 255) / 256, 256, 0, stream>>>(x, xh, n);
  n = H_ * C_ * 3 * DK_;
  k_pack_wqkv_T<<<(n + 255) / 256, 256, 0, stream>>>(Wqkv, WqkvT);
  n = C_ * C_;
  k_f32_to_f16_T<<<(n + 255) / 256, 256, 0, stream>>>(WO, WOT, C_, C_);
  n = C_ * FF_;
  k_f32_to_f16_T<<<(n + 255) / 256, 256, 0, stream>>>(W1, W1T, C_, FF_);
  n = FF_ * C_;
  k_f32_to_f16_T<<<(n + 255) / 256, 256, 0, stream>>>(W2, W2T, FF_, C_);

  // 1) fused QKV projection (routes Q/K/V per head; Q pre-scaled, V transposed)
  k_gemm<0><<<dim3(QKVW / BN, NROW / BM), 256, 0, stream>>>(
      xh, WqkvT, NROW, QKVW, C_, bqkv, nullptr, nullptr, nullptr, Qh, Kh, Vth);
  // 2) causal flash attention
  k_attn<<<dim3(T_ / 64, B_ * H_), 128, 0, stream>>>(Qh, Kh, Vth, attnout);
  // 3) output projection + bias + residual(x) -> y1 (f32)
  k_gemm<2><<<dim3(C_ / BN, NROW / BM), 256, 0, stream>>>(
      attnout, WOT, NROW, C_, C_, bO, x, y1, nullptr, nullptr, nullptr, nullptr);
  // 4) LN1 -> x1 (f32 residual) + x1h (f16 GEMM input)
  k_layernorm<<<NROW, 256, 0, stream>>>(y1, g1, be1, x1, x1h);
  // 5) FFN1: x1h @ W1 + b1, ReLU -> h1h (f16)
  k_gemm<1><<<dim3(FF_ / BN, NROW / BM), 256, 0, stream>>>(
      x1h, W1T, NROW, FF_, C_, b1, nullptr, nullptr, h1h, nullptr, nullptr, nullptr);
  // 6) FFN2: h1h @ W2 + b2 + residual(x1) -> y2 (f32)
  k_gemm<2><<<dim3(C_ / BN, NROW / BM), 256, 0, stream>>>(
      h1h, W2T, NROW, C_, FF_, b2, x1, y2, nullptr, nullptr, nullptr, nullptr);
  // 7) LN2 -> final output (f32)
  k_layernorm<<<NROW, 256, 0, stream>>>(y2, g2, be2, out, nullptr);
}
